// HybridBlock_69887707841220
// MI455X (gfx1250) — matrix operand order repaired and checked by reading, not compiled, not run on hardware
//
#include <hip/hip_runtime.h>
#include <math.h>

// ---------------------------------------------------------------------------
// Types for CDNA5 WMMA
// ---------------------------------------------------------------------------
typedef __attribute__((ext_vector_type(16))) __bf16 v16bf;
typedef __attribute__((ext_vector_type(8)))  float  v8f;

#define HID   1024
#define STATE 64
#define INTER 4096
#define BATCH 4
#define SEQ   2048
#define RTOT  8192          // BATCH*SEQ
#define NCH   32            // chunks per batch (SEQ/64)
#define LSTR  40            // LDS row stride in halves (32 + 8 pad) -> 80B, conflict-free

union Frag16 { uint4 q[2]; v16bf v; };

// ---------------------------------------------------------------------------
// CDNA5 async global->LDS copy (ASYNCcnt-tracked), 16B per lane per issue.
// LDS operand: flat address truncated to 32 bits == wave-relative LDS offset.
// ---------------------------------------------------------------------------
__device__ inline void async_copy16(void* lds_p, const void* g_p) {
    unsigned loff = (unsigned)(unsigned long long)lds_p;
    asm volatile("global_load_async_to_lds_b128 %0, %1, off"
                 :: "v"(loff), "v"(g_p) : "memory");
}

template<int N>
__device__ inline void wait_asynccnt() {
    asm volatile("s_wait_asynccnt %0" :: "n"(N) : "memory");
}

// Issue one ROWS x 32 bf16 tile (row-major, leading dim ldg) into LDS, async.
// 256 threads; ROWS*4 16B-chunks; exact multiple of 256 -> branch-free.
template<int ROWS>
__device__ inline void issue_tile(__bf16* lds, const __bf16* g, long ldg,
                                  long rowBase, int k0, int tid) {
    constexpr int IT = (ROWS * 4) / 256;    // chunks per thread
#pragma unroll
    for (int it = 0; it < IT; it++) {
        int ch = it * 256 + tid;
        int r  = ch >> 2;
        int cc = (ch & 3) * 8;
        async_copy16(lds + r * LSTR + cc, g + (rowBase + r) * ldg + k0 + cc);
    }
}

// A fragment: 16x32 bf16. lane<16: row=lane, K slots {0..7,16..23}; lane>=16: row=lane-16, K {8..15,24..31}
__device__ inline v16bf load_afrag(const __bf16* lds, int row16, int lane) {
    int m  = row16 + (lane & 15);
    int hi = lane >> 4;
    const __bf16* p = lds + m * LSTR + hi * 8;
    Frag16 f;
    f.q[0] = *(const uint4*)(p);
    f.q[1] = *(const uint4*)(p + 16);
    return f.v;
}

// B fragment: 32x16 bf16 (W stored N-major rows of K). lane<16: n=lane, K {0..7,16..23}; lane>=16: n=lane-16, K {8..15,24..31}
__device__ inline v16bf load_bfrag(const __bf16* lds, int col16, int lane) {
    int n  = col16 + (lane & 15);
    int hi = lane >> 4;
    const __bf16* p = lds + n * LSTR + hi * 8;
    Frag16 f;
    f.q[0] = *(const uint4*)(p);
    f.q[1] = *(const uint4*)(p + 16);
    return f.v;
}

// ---------------------------------------------------------------------------
// Double-buffered async block-tile GEMM core:
// Out(M,N) = A(M,K) @ W(N,K)^T, bf16 in, f32 accumulate.
// 256 threads = 8 waves, wave tile 32x64 -> acc[2][4] of 16x16.
// ---------------------------------------------------------------------------
template<int BM, int BN, int WMW>
__device__ inline void gemm_core(const __bf16* Ag, long lda,
                                 const __bf16* Wg, long ldw, int K,
                                 long mBase, long nBase,
                                 __bf16* ldsA, __bf16* ldsW,   // each 2*B?*LSTR
                                 v8f acc[2][4]) {
    const int tid  = threadIdx.x;
    const int lane = tid & 31;
    const int wid  = tid >> 5;
    const int wm   = (wid % WMW) * 32;
    const int wn   = (wid / WMW) * 64;
    constexpr int IPT = (BM * 4) / 256 + (BN * 4) / 256;   // async issues per thread per k-step

#pragma unroll
    for (int i = 0; i < 2; i++)
#pragma unroll
        for (int j = 0; j < 4; j++)
            acc[i][j] = (v8f){0.f,0.f,0.f,0.f,0.f,0.f,0.f,0.f};

    // prologue: stage first tiles into buffer 0
    issue_tile<BM>(ldsA, Ag, lda, mBase, 0, tid);
    issue_tile<BN>(ldsW, Wg, ldw, nBase, 0, tid);

    int cur = 0;
    for (int k0 = 0; k0 < K; k0 += 32) {
        const bool more = (k0 + 32 < K);
        if (more) {     // stage next tiles into the other buffer (uniform branch)
            int nxt = cur ^ 1;
            issue_tile<BM>(ldsA + nxt * (BM * LSTR), Ag, lda, mBase, k0 + 32, tid);
            issue_tile<BN>(ldsW + nxt * (BN * LSTR), Wg, ldw, nBase, k0 + 32, tid);
            wait_asynccnt<IPT>();     // in-order completion: current tiles resident
        } else {
            wait_asynccnt<0>();
        }
        __syncthreads();              // all waves' copies for current buffer done

        const __bf16* bufA = ldsA + cur * (BM * LSTR);
        const __bf16* bufW = ldsW + cur * (BN * LSTR);
        v16bf a0 = load_afrag(bufA, wm,      lane);
        v16bf a1 = load_afrag(bufA, wm + 16, lane);
#pragma unroll
        for (int j = 0; j < 4; j++) {
            v16bf b = load_bfrag(bufW, wn + j * 16, lane);
            acc[0][j] = __builtin_amdgcn_wmma_f32_16x16x32_bf16(
                false, a0, false, b, (short)0, acc[0][j], false, false);
            acc[1][j] = __builtin_amdgcn_wmma_f32_16x16x32_bf16(
                false, a1, false, b, (short)0, acc[1][j], false, false);
        }
        __syncthreads();              // done reading buf[cur]; safe to overwrite next iter
        cur ^= 1;
    }
}

// ---------------------------------------------------------------------------
// Kernels
// ---------------------------------------------------------------------------

// Convert fp32 weights to bf16 (w_gate, w_up, w_down, C)
__global__ void convert_kernel(const float* wg, const float* wu, const float* wd,
                               const float* C,
                               __bf16* wg16, __bf16* wu16, __bf16* wd16, __bf16* c16) {
    const long n1 = (long)INTER * HID;
    const long total = 3 * n1 + (long)HID * STATE;
    for (long idx = (long)blockIdx.x * 256 + threadIdx.x; idx < total;
         idx += (long)gridDim.x * 256) {
        if (idx < n1)            wg16[idx]          = (__bf16)wg[idx];
        else if (idx < 2 * n1)   wu16[idx - n1]     = (__bf16)wu[idx - n1];
        else if (idx < 3 * n1)   wd16[idx - 2 * n1] = (__bf16)wd[idx - 2 * n1];
        else                     c16[idx - 3 * n1]  = (__bf16)C[idx - 3 * n1];
    }
}

// dt = exp(log_dt); A_bar = expm(A*mean(dt)) via Taylor; Apow = A_bar^64; B_bar bf16
__global__ void prep_kernel(const float* A, const float* B, const float* log_dt,
                            float* Abar, float* Apow, float* dtbuf, __bf16* bbar16) {
    __shared__ float Msh[64][65];
    __shared__ float T0[64][65];
    __shared__ float T1[64][65];
    __shared__ float red[256];
    __shared__ float meansh;
    const int tid = threadIdx.x;

    float s = 0.f;
    for (int i = tid; i < HID; i += 256) {
        float d = expf(log_dt[i]);
        dtbuf[i] = d;
        s += d;
    }
    red[tid] = s; __syncthreads();
    for (int st = 128; st > 0; st >>= 1) {
        if (tid < st) red[tid] += red[tid + st];
        __syncthreads();
    }
    if (tid == 0) meansh = red[0] / (float)HID;
    __syncthreads();
    const float mdt = meansh;

    for (int e = tid; e < 4096; e += 256) Msh[e >> 6][e & 63] = A[e] * mdt;
    __syncthreads();

    // E = I + M; T = M; for k=2..8: T = T@M/k; E += T
    float E[16];
#pragma unroll
    for (int u = 0; u < 16; u++) {
        int e = tid * 16 + u, i = e >> 6, j = e & 63;
        E[u] = Msh[i][j] + ((i == j) ? 1.f : 0.f);
        T0[i][j] = Msh[i][j];
    }
    __syncthreads();
    for (int k = 2; k <= 8; k++) {
        float invk = 1.f / (float)k;
#pragma unroll
        for (int u = 0; u < 16; u++) {
            int e = tid * 16 + u, i = e >> 6, j = e & 63;
            float acc = 0.f;
            for (int l = 0; l < 64; l++) acc += T0[i][l] * Msh[l][j];
            acc *= invk;
            T1[i][j] = acc;
            E[u] += acc;
        }
        __syncthreads();
#pragma unroll
        for (int u = 0; u < 16; u++) { int e = tid * 16 + u; T0[e >> 6][e & 63] = T1[e >> 6][e & 63]; }
        __syncthreads();
    }
#pragma unroll
    for (int u = 0; u < 16; u++) {
        int e = tid * 16 + u;
        Abar[e] = E[u];
        T0[e >> 6][e & 63] = E[u];
    }
    __syncthreads();
    // 6 squarings -> A_bar^64
    for (int q = 0; q < 6; q++) {
#pragma unroll
        for (int u = 0; u < 16; u++) {
            int e = tid * 16 + u, i = e >> 6, j = e & 63;
            float acc = 0.f;
            for (int l = 0; l < 64; l++) acc += T0[i][l] * T0[l][j];
            T1[i][j] = acc;
        }
        __syncthreads();
#pragma unroll
        for (int u = 0; u < 16; u++) { int e = tid * 16 + u; T0[e >> 6][e & 63] = T1[e >> 6][e & 63]; }
        __syncthreads();
    }
#pragma unroll
    for (int u = 0; u < 16; u++) { int e = tid * 16 + u; Apow[e] = T0[e >> 6][e & 63]; }

    // B_bar = B * dt (row s, col h), store bf16
    for (int e = tid; e < STATE * HID; e += 256) {
        int h = e & (HID - 1);
        bbar16[e] = (__bf16)(B[e] * dtbuf[h]);
    }
}

// RMSNorm per row; optionally keep fp32 copy, always emit bf16 copy
__global__ void rmsnorm_kernel(const float* x, const float* w,
                               float* out32, __bf16* out16) {
    const int tid = threadIdx.x;
    const long row = blockIdx.x;
    const float* xr = x + row * HID;
    float ss = 0.f;
    for (int i = tid; i < HID; i += 256) { float v = xr[i]; ss += v * v; }
    __shared__ float red[256];
    red[tid] = ss; __syncthreads();
    for (int s = 128; s > 0; s >>= 1) {
        if (tid < s) red[tid] += red[tid + s];
        __syncthreads();
    }
    const float scale = rsqrtf(red[0] / (float)HID + 1e-6f);
    for (int i = tid; i < HID; i += 256) {
        float v = xr[i] * scale * w[i];
        if (out32) out32[row * HID + i] = v;
        out16[row * HID + i] = (__bf16)v;
    }
}

// U = h16 @ B_bar^T : (8192,1024)x(64,1024)^T -> (8192,64) fp32
__global__ __launch_bounds__(256) void gemm_u_kernel(const __bf16* h16,
                                                     const __bf16* bbar16,
                                                     float* U) {
    __shared__ __attribute__((aligned(16))) __bf16 ldsA[2 * 256 * LSTR];
    __shared__ __attribute__((aligned(16))) __bf16 ldsW[2 * 64 * LSTR];
    v8f acc[2][4];
    const long mBase = (long)blockIdx.x * 256;
    gemm_core<256, 64, 8>(h16, HID, bbar16, HID, HID, mBase, 0, ldsA, ldsW, acc);

    const int tid = threadIdx.x, lane = tid & 31, wid = tid >> 5;
    const int wm = (wid % 8) * 32;
    const int rOff = (lane >> 4) * 8, cOff = lane & 15;
#pragma unroll
    for (int i = 0; i < 2; i++)
#pragma unroll
        for (int j = 0; j < 4; j++) {
            long row0 = mBase + wm + i * 16 + rOff;
            long col  = j * 16 + cOff;
#pragma unroll
            for (int r = 0; r < 8; r++)
                U[(row0 + r) * STATE + col] = acc[i][j][r];
        }
}

// Pass A: per-chunk local scan with zero carry. grid 128 blocks x 64 threads.
__global__ void scan_local_kernel(const float* Abar, const float* U,
                                  float* S32, float* Vbuf) {
    const int blk = blockIdx.x;
    const int b = blk >> 5, c = blk & 31;
    const int i = threadIdx.x;
    __shared__ float st[64];
    float arow[64];
#pragma unroll
    for (int j = 0; j < 64; j++) arow[j] = Abar[i * 64 + j];
    st[i] = 0.f;
    float s = 0.f;
    const long base = ((long)b * SEQ + c * 64) * STATE;
    __syncthreads();
    for (int t = 0; t < 64; t++) {
        float acc = U[base + t * STATE + i];
        for (int j = 0; j < 64; j++) acc += arow[j] * st[j];
        __syncthreads();
        st[i] = acc; s = acc;
        S32[base + t * STATE + i] = acc;
        __syncthreads();
    }
    Vbuf[(long)(b * NCH + c) * STATE + i] = s;
}

// Pass B: inter-chunk carry recurrence using A_bar^64. grid 4 blocks x 64 threads.
__global__ void scan_carry_kernel(const float* Apow, const float* Vbuf, float* Carry) {
    const int b = blockIdx.x;
    const int i = threadIdx.x;
    __shared__ float st[64];
    float arow[64];
#pragma unroll
    for (int j = 0; j < 64; j++) arow[j] = Apow[i * 64 + j];
    float carry = 0.f;
    st[i] = 0.f;
    __syncthreads();
    for (int c = 0; c < NCH; c++) {
        Carry[(long)(b * NCH + c) * STATE + i] = carry;
        float acc = Vbuf[(long)(b * NCH + c) * STATE + i];
        for (int j = 0; j < 64; j++) acc += arow[j] * st[j];
        __syncthreads();
        st[i] = acc; carry = acc;
        __syncthreads();
    }
}

// Pass C: s_t = w_t + A_bar^t * carry ; emit bf16 states.
__global__ void scan_fix_kernel(const float* Abar, const float* Carry,
                                const float* S32, __bf16* S16) {
    const int blk = blockIdx.x;
    const int b = blk >> 5, c = blk & 31;
    const int i = threadIdx.x;
    __shared__ float q[64];
    float arow[64];
#pragma unroll
    for (int j = 0; j < 64; j++) arow[j] = Abar[i * 64 + j];
    q[i] = Carry[(long)(b * NCH + c) * STATE + i];
    const long base = ((long)b * SEQ + c * 64) * STATE;
    __syncthreads();
    for (int t = 0; t < 64; t++) {
        float acc = 0.f;
        for (int j = 0; j < 64; j++) acc += arow[j] * q[j];
        float sfin = S32[base + t * STATE + i] + acc;
        S16[base + t * STATE + i] = (__bf16)sfin;
        __syncthreads();
        q[i] = acc;
        __syncthreads();
    }
}

// h2 = x + S@C^T + D*h_norm  -> d_out (fp32)
__global__ __launch_bounds__(256) void gemm_y_kernel(const __bf16* S16, const __bf16* c16,
                                                     const float* h32, const float* x,
                                                     const float* Dv, float* out) {
    __shared__ __attribute__((aligned(16))) __bf16 ldsA[2 * 128 * LSTR];
    __shared__ __attribute__((aligned(16))) __bf16 ldsW[2 * 128 * LSTR];
    v8f acc[2][4];
    const long mBase = (long)blockIdx.x * 128;
    const long nBase = (long)blockIdx.y * 128;
    gemm_core<128, 128, 4>(S16, STATE, c16, STATE, STATE, mBase, nBase, ldsA, ldsW, acc);

    const int tid = threadIdx.x, lane = tid & 31, wid = tid >> 5;
    const int wm = (wid % 4) * 32, wn = (wid / 4) * 64;
    const int rOff = (lane >> 4) * 8, cOff = lane & 15;
#pragma unroll
    for (int i = 0; i < 2; i++)
#pragma unroll
        for (int j = 0; j < 4; j++) {
            long row0 = mBase + wm + i * 16 + rOff;
            long col  = nBase + wn + j * 16 + cOff;
            float d = Dv[col];
#pragma unroll
            for (int r = 0; r < 8; r++) {
                long idx = (row0 + r) * HID + col;
                out[idx] = x[idx] + acc[i][j][r] + d * h32[idx];
            }
        }
}

// act = silu(hn@wg^T) * (hn@wu^T), bf16 out
__global__ __launch_bounds__(256) void gemm_gateup_kernel(const __bf16* hn16,
                                                          const __bf16* wg16,
                                                          const __bf16* wu16,
                                                          __bf16* act16) {
    __shared__ __attribute__((aligned(16))) __bf16 ldsA[2 * 128 * LSTR];
    __shared__ __attribute__((aligned(16))) __bf16 ldsW[2 * 128 * LSTR];
    v8f accg[2][4], accu[2][4];
    const long mBase = (long)blockIdx.x * 128;
    const long nBase = (long)blockIdx.y * 128;
    gemm_core<128, 128, 4>(hn16, HID, wg16, HID, HID, mBase, nBase, ldsA, ldsW, accg);
    gemm_core<128, 128, 4>(hn16, HID, wu16, HID, HID, mBase, nBase, ldsA, ldsW, accu);

    const int tid = threadIdx.x, lane = tid & 31, wid = tid >> 5;
    const int wm = (wid % 4) * 32, wn = (wid / 4) * 64;
    const int rOff = (lane >> 4) * 8, cOff = lane & 15;
#pragma unroll
    for (int i = 0; i < 2; i++)
#pragma unroll
        for (int j = 0; j < 4; j++) {
            long row0 = mBase + wm + i * 16 + rOff;
            long col  = nBase + wn + j * 16 + cOff;
#pragma unroll
            for (int r = 0; r < 8; r++) {
                float g = accg[i][j][r];
                float u = accu[i][j][r];
                float a = (g / (1.f + expf(-g))) * u;
                act16[(row0 + r) * INTER + col] = (__bf16)a;
            }
        }
}

// out += act @ wd^T (out already holds h2 residual)
__global__ __launch_bounds__(256) void gemm_down_kernel(const __bf16* act16,
                                                        const __bf16* wd16,
                                                        float* out) {
    __shared__ __attribute__((aligned(16))) __bf16 ldsA[2 * 128 * LSTR];
    __shared__ __attribute__((aligned(16))) __bf16 ldsW[2 * 128 * LSTR];
    v8f acc[2][4];
    const long mBase = (long)blockIdx.x * 128;
    const long nBase = (long)blockIdx.y * 128;
    gemm_core<128, 128, 4>(act16, INTER, wd16, INTER, INTER, mBase, nBase, ldsA, ldsW, acc);

    const int tid = threadIdx.x, lane = tid & 31, wid = tid >> 5;
    const int wm = (wid % 4) * 32, wn = (wid / 4) * 64;
    const int rOff = (lane >> 4) * 8, cOff = lane & 15;
#pragma unroll
    for (int i = 0; i < 2; i++)
#pragma unroll
        for (int j = 0; j < 4; j++) {
            long row0 = mBase + wm + i * 16 + rOff;
            long col  = nBase + wn + j * 16 + cOff;
#pragma unroll
            for (int r = 0; r < 8; r++) {
                long idx = (row0 + r) * HID + col;
                out[idx] = out[idx] + acc[i][j][r];
            }
        }
}

// ---------------------------------------------------------------------------
// Host launcher
// ---------------------------------------------------------------------------
extern "C" void kernel_launch(void* const* d_in, const int* in_sizes, int n_in,
                              void* d_out, int out_size, void* d_ws, size_t ws_size,
                              hipStream_t stream) {
    const float* x      = (const float*)d_in[0];   // (4,2048,1024)
    const float* A      = (const float*)d_in[1];   // (64,64)
    const float* B      = (const float*)d_in[2];   // (64,1024)
    const float* C      = (const float*)d_in[3];   // (1024,64)
    const float* Dv     = (const float*)d_in[4];   // (1024,)
    const float* log_dt = (const float*)d_in[5];   // (1024,)
    const float* wg     = (const float*)d_in[6];   // (4096,1024)
    const float* wu     = (const float*)d_in[7];   // (4096,1024)
    const float* wd     = (const float*)d_in[8];   // (1024,4096)
    const float* ln1    = (const float*)d_in[9];
    const float* ln2    = (const float*)d_in[10];
    float* out = (float*)d_out;

    // workspace carve-out
    char* p = (char*)d_ws;
    auto alloc = [&](size_t bytes) -> void* {
        void* r = (void*)p;
        p += (bytes + 255) & ~(size_t)255;
        return r;
    };
    float*  h32    = (float*) alloc((size_t)RTOT * HID * 4);
    __bf16* h16    = (__bf16*)alloc((size_t)RTOT * HID * 2);   // reused as hn16
    __bf16* act16  = (__bf16*)alloc((size_t)RTOT * INTER * 2);
    __bf16* wg16   = (__bf16*)alloc((size_t)INTER * HID * 2);
    __bf16* wu16   = (__bf16*)alloc((size_t)INTER * HID * 2);
    __bf16* wd16   = (__bf16*)alloc((size_t)HID * INTER * 2);
    __bf16* c16    = (__bf16*)alloc((size_t)HID * STATE * 2);
    __bf16* bbar16 = (__bf16*)alloc((size_t)STATE * HID * 2);
    float*  U      = (float*) alloc((size_t)RTOT * STATE * 4);
    float*  S32    = (float*) alloc((size_t)RTOT * STATE * 4);
    __bf16* S16    = (__bf16*)alloc((size_t)RTOT * STATE * 2);
    float*  Vbuf   = (float*) alloc((size_t)BATCH * NCH * STATE * 4);
    float*  Carry  = (float*) alloc((size_t)BATCH * NCH * STATE * 4);
    float*  Abar   = (float*) alloc(64 * 64 * 4);
    float*  Apow   = (float*) alloc(64 * 64 * 4);
    float*  dtbuf  = (float*) alloc(HID * 4);

    // 1) weights -> bf16 ; 2) dt/expm/B_bar prep
    convert_kernel<<<4096, 256, 0, stream>>>(wg, wu, wd, C, wg16, wu16, wd16, c16);
    prep_kernel<<<1, 256, 0, stream>>>(A, B, log_dt, Abar, Apow, dtbuf, bbar16);

    // 3) h = rmsnorm(x, ln1)
    rmsnorm_kernel<<<RTOT, 256, 0, stream>>>(x, ln1, h32, h16);

    // 4) U = h @ B_bar^T
    gemm_u_kernel<<<32, 256, 0, stream>>>(h16, bbar16, U);

    // 5) chunked linear scan
    scan_local_kernel<<<BATCH * NCH, 64, 0, stream>>>(Abar, U, S32, Vbuf);
    scan_carry_kernel<<<BATCH, 64, 0, stream>>>(Apow, Vbuf, Carry);
    scan_fix_kernel<<<BATCH * NCH, 64, 0, stream>>>(Abar, Carry, S32, S16);

    // 6) h2 = x + S@C^T + D*h   (into d_out)
    gemm_y_kernel<<<dim3(RTOT / 128, HID / 128), 256, 0, stream>>>(S16, c16, h32, x, Dv, out);

    // 7) hn = rmsnorm(h2, ln2) -> bf16 (reuse h16)
    rmsnorm_kernel<<<RTOT, 256, 0, stream>>>(out, ln2, (float*)nullptr, h16);

    // 8) act = silu(hn@wg^T) * (hn@wu^T)
    gemm_gateup_kernel<<<dim3(RTOT / 128, INTER / 128), 256, 0, stream>>>(h16, wg16, wu16, act16);

    // 9) out = h2 + act @ wd^T
    gemm_down_kernel<<<dim3(RTOT / 128, HID / 128), 256, 0, stream>>>(act16, wd16, out);
}
